// UniTransformerO2TwoUpdateGeneral_1692217115463
// MI455X (gfx1250) — compile-verified, matrix operand order checked by reading, not run on hardware
//
#include <hip/hip_runtime.h>
#include <hip/hip_bf16.h>

// ---------------------------------------------------------------------------
// UniTransformerO2TwoUpdateGeneral for MI455X (gfx1250, wave32, WMMA).
// Edge MLPs use 64-row tiles with 4-way B-fragment reuse per wave to raise
// arithmetic intensity on the v_wmma_f32_16x16x32_bf16 path.
// ---------------------------------------------------------------------------

#define HID   128
#define HEADS 16
#define HD    8
#define NG    20
#define EDIM  4
#define RFEAT 80
#define KVDIM 340
#define KVPAD 352   // 11 chunks of 32
#define ETILE 64    // edges per block
#define RB    4     // 16-row blocks per edge tile

typedef __attribute__((ext_vector_type(16))) __bf16 v16bf;
typedef __attribute__((ext_vector_type(8)))  float  v8f;

// ---------------- small helpers ----------------
__device__ __forceinline__ unsigned short bfbits(float x) {
  unsigned u = __float_as_uint(x);
  return (unsigned short)((u + 0x7FFFu + ((u >> 16) & 1u)) >> 16);
}
__device__ __forceinline__ __bf16 f2bf(float x) {
  union { unsigned short s; __bf16 b; } o; o.s = bfbits(x); return o.b;
}
__device__ __forceinline__ float bf2f(unsigned short s) {
  return __uint_as_float(((unsigned)s) << 16);
}
// order-preserving float<->uint for atomicMax
__device__ __forceinline__ unsigned encf(float f) {
  unsigned u = __float_as_uint(f);
  return (u & 0x80000000u) ? ~u : (u | 0x80000000u);
}
__device__ __forceinline__ float decf(unsigned u) {
  return (u & 0x80000000u) ? __uint_as_float(u & 0x7FFFFFFFu) : __uint_as_float(~u);
}
#define MX_INIT 0x00800000u  // encf(-FLT_MAX)

// ---------------- WMMA fragment loaders ----------------
// A: 16x32 bf16 tile, row-major in LDS (row stride = astride elements, even).
__device__ __forceinline__ v16bf load_afrag(const __bf16* A, int astride,
                                            int kbase, int lane) {
  union { v16bf v; unsigned u[8]; } r;
  int row = lane & 15, half = lane >> 4;
  const __bf16* base = A + row * astride + kbase + half * 8;
#pragma unroll
  for (int v = 0; v < 8; ++v) {
    int kk = (v < 4) ? (v * 2) : (16 + (v - 4) * 2);
    r.u[v] = *(const unsigned*)(base + kk);
  }
  return r.v;
}
// B fragments pre-packed: [chunk][tile][lane][8 dwords]
__device__ __forceinline__ v16bf load_bfrag(const unsigned* wp, int chunk,
                                            int tile, int douttiles, int lane) {
  const unsigned* p = wp + (((size_t)chunk * douttiles + tile) * 32 + lane) * 8;
  union { v16bf v; unsigned u[8]; } r;
#pragma unroll
  for (int v = 0; v < 8; ++v) r.u[v] = p[v];
  return r.v;
}

// ---------------- MLP descriptor ----------------
struct MLPW {
  const unsigned* w1p;
  const unsigned* w2p;
  const float *b1, *g, *be, *b2;
  int chunks1;           // ceil(din/32)
  int douttiles;         // dout/16
};

// ======== 16-row MLP tile (node-side, small N) ========
__device__ __forceinline__ void mlp_tile(const __bf16* A, int astride,
                                         const MLPW w, float* midf,
                                         __bf16* midb, float* outf, int tid) {
  const int lane = tid & 31, wv = tid >> 5;
  for (int t = wv; t < 8; t += 4) {
    v8f acc = {};
    for (int c = 0; c < w.chunks1; ++c) {
      v16bf a = load_afrag(A, astride, c * 32, lane);
      v16bf b = load_bfrag(w.w1p, c, t, 8, lane);
      acc = __builtin_amdgcn_wmma_f32_16x16x32_bf16(false, a, false, b,
                                                    (short)0, acc, false, false);
    }
    int col = t * 16 + (lane & 15), half = lane >> 4;
#pragma unroll
    for (int r = 0; r < 8; ++r)
      midf[(r + half * 8) * HID + col] = acc[r] + w.b1[col];
  }
  __syncthreads();
  if (tid < 16) {
    float mu = 0.f;
    for (int c = 0; c < HID; ++c) mu += midf[tid * HID + c];
    mu *= (1.f / HID);
    float var = 0.f;
    for (int c = 0; c < HID; ++c) { float d = midf[tid * HID + c] - mu; var += d * d; }
    var *= (1.f / HID);
    float rstd = rsqrtf(var + 1e-5f);
    for (int c = 0; c < HID; ++c) {
      float hh = (midf[tid * HID + c] - mu) * rstd * w.g[c] + w.be[c];
      midb[tid * HID + c] = f2bf(fmaxf(hh, 0.f));
    }
  }
  __syncthreads();
  const int dout = w.douttiles * 16;
  for (int t = wv; t < w.douttiles; t += 4) {
    v8f acc = {};
#pragma unroll
    for (int c = 0; c < 4; ++c) {
      v16bf a = load_afrag(midb, HID, c * 32, lane);
      v16bf b = load_bfrag(w.w2p, c, t, w.douttiles, lane);
      acc = __builtin_amdgcn_wmma_f32_16x16x32_bf16(false, a, false, b,
                                                    (short)0, acc, false, false);
    }
    int col = t * 16 + (lane & 15), half = lane >> 4;
#pragma unroll
    for (int r = 0; r < 8; ++r)
      outf[(r + half * 8) * dout + col] = acc[r] + w.b2[col];
  }
  __syncthreads();
}

// ======== 64-row MLP tile (edge-side): 4x B-fragment reuse ========
// mid (f32 [64][128]) holds layer-1 pre-LN output, then is overwritten with
// the final layer-2 output (stride dout).
__device__ __forceinline__ void mlp_tile64(const __bf16* A, int astride,
                                           const MLPW w, float* mid,
                                           __bf16* midb, int tid) {
  const int lane = tid & 31, wv = tid >> 5;
  // ---- layer 1 ----
  for (int t = wv; t < 8; t += 4) {
    v8f acc[RB];
    v8f zero = {};
#pragma unroll
    for (int m = 0; m < RB; ++m) acc[m] = zero;
    for (int c = 0; c < w.chunks1; ++c) {
      v16bf b = load_bfrag(w.w1p, c, t, 8, lane);
#pragma unroll
      for (int m = 0; m < RB; ++m) {
        v16bf a = load_afrag(A + m * 16 * astride, astride, c * 32, lane);
        acc[m] = __builtin_amdgcn_wmma_f32_16x16x32_bf16(false, a, false, b,
                                                         (short)0, acc[m], false, false);
      }
    }
    int col = t * 16 + (lane & 15), half = lane >> 4;
#pragma unroll
    for (int m = 0; m < RB; ++m)
#pragma unroll
      for (int r = 0; r < 8; ++r)
        mid[(m * 16 + r + half * 8) * HID + col] = acc[m][r] + w.b1[col];
  }
  __syncthreads();
  // ---- LayerNorm + ReLU (one thread per row) ----
  if (tid < ETILE) {
    float mu = 0.f;
    for (int c = 0; c < HID; ++c) mu += mid[tid * HID + c];
    mu *= (1.f / HID);
    float var = 0.f;
    for (int c = 0; c < HID; ++c) { float d = mid[tid * HID + c] - mu; var += d * d; }
    var *= (1.f / HID);
    float rstd = rsqrtf(var + 1e-5f);
    for (int c = 0; c < HID; ++c) {
      float hh = (mid[tid * HID + c] - mu) * rstd * w.g[c] + w.be[c];
      midb[tid * HID + c] = f2bf(fmaxf(hh, 0.f));
    }
  }
  __syncthreads();
  // ---- layer 2 (overwrites mid with stride dout) ----
  const int dout = w.douttiles * 16;
  for (int t = wv; t < w.douttiles; t += 4) {
    v8f acc[RB];
    v8f zero = {};
#pragma unroll
    for (int m = 0; m < RB; ++m) acc[m] = zero;
#pragma unroll
    for (int c = 0; c < 4; ++c) {
      v16bf b = load_bfrag(w.w2p, c, t, w.douttiles, lane);
#pragma unroll
      for (int m = 0; m < RB; ++m) {
        v16bf a = load_afrag(midb + m * 16 * HID, HID, c * 32, lane);
        acc[m] = __builtin_amdgcn_wmma_f32_16x16x32_bf16(false, a, false, b,
                                                         (short)0, acc[m], false, false);
      }
    }
    int col = t * 16 + (lane & 15), half = lane >> 4;
#pragma unroll
    for (int m = 0; m < RB; ++m)
#pragma unroll
      for (int r = 0; r < 8; ++r)
        mid[(m * 16 + r + half * 8) * dout + col] = acc[m][r] + w.b2[col];
  }
  __syncthreads();
}

// ---------------- weight packer ----------------
__global__ void pack_weight(const float* __restrict__ w, int din, int dout,
                            unsigned* __restrict__ dst, int chunks) {
  int tiles = dout >> 4;
  int total = chunks * tiles * 32 * 8;
  for (int i = blockIdx.x * blockDim.x + threadIdx.x; i < total;
       i += gridDim.x * blockDim.x) {
    int v = i & 7, lane = (i >> 3) & 31;
    int rem = i >> 8, t = rem % tiles, c = rem / tiles;
    int col = t * 16 + (lane & 15), half = lane >> 4;
    int kk = ((v < 4) ? v * 2 : 16 + (v - 4) * 2) + half * 8 + c * 32;
    float a0 = (kk     < din) ? w[(size_t)kk * dout + col]       : 0.f;
    float a1 = (kk + 1 < din) ? w[(size_t)(kk + 1) * dout + col] : 0.f;
    dst[i] = (unsigned)bfbits(a0) | ((unsigned)bfbits(a1) << 16);
  }
}

__global__ void fill_u32(unsigned* p, unsigned v, long long n) {
  long long i = (long long)blockIdx.x * blockDim.x + threadIdx.x;
  if (i < n) p[i] = v;
}

// ---------------- node MLP (q / out_fc) ----------------
__global__ __launch_bounds__(128)
void node_mlp(const float* __restrict__ A0, const float* __restrict__ A1,
              MLPW w, const float* __restrict__ resid,
              float* __restrict__ out, int N) {
  __shared__ __bf16 at[16 * 256];
  __shared__ float  midf[16 * HID];
  __shared__ __bf16 midb[16 * HID];
  __shared__ float  outf[16 * HID];
  const int tid = threadIdx.x, r0 = blockIdx.x * 16;
  const int din = A1 ? 256 : 128;
  for (int idx = tid; idx < 16 * din; idx += 128) {
    int r = idx / din, c = idx % din, gr = r0 + r;
    float v = 0.f;
    if (gr < N) v = (c < 128) ? A0[(size_t)gr * 128 + c]
                              : A1[(size_t)gr * 128 + (c - 128)];
    at[r * din + c] = f2bf(v);
  }
  __syncthreads();
  mlp_tile(at, din, w, midf, midb, outf, tid);
  const int dout = w.douttiles * 16;
  for (int idx = tid; idx < 16 * dout; idx += 128) {
    int r = idx / dout, c = idx % dout, gr = r0 + r;
    if (gr < N) {
      float v = outf[idx];
      if (resid) v += resid[(size_t)gr * 128 + c];
      out[(size_t)gr * dout + c] = v;
    }
  }
}

// ---------------- fusion: concat[h_in, pad] @ fusion_w + b ----------------
__global__ __launch_bounds__(128)
void fusion_kernel(const float* __restrict__ hin, const float* __restrict__ pad,
                   const unsigned* __restrict__ wp, const float* __restrict__ bias,
                   float* __restrict__ out, int N) {
  __shared__ __bf16 at[16 * 256];
  __shared__ float  outf[16 * HID];
  const int tid = threadIdx.x, r0 = blockIdx.x * 16;
  for (int idx = tid; idx < 16 * 256; idx += 128) {
    int r = idx >> 8, c = idx & 255, gr = r0 + r;
    float v = 0.f;
    if (gr < N) v = (c < 128) ? hin[(size_t)gr * 128 + c]
                              : pad[(size_t)gr * 128 + (c - 128)];
    at[idx] = f2bf(v);
  }
  __syncthreads();
  const int lane = tid & 31, wv = tid >> 5;
  for (int t = wv; t < 8; t += 4) {
    v8f acc = {};
#pragma unroll
    for (int c = 0; c < 8; ++c) {
      v16bf a = load_afrag(at, 256, c * 32, lane);
      v16bf b = load_bfrag(wp, c, t, 8, lane);
      acc = __builtin_amdgcn_wmma_f32_16x16x32_bf16(false, a, false, b,
                                                    (short)0, acc, false, false);
    }
    int col = t * 16 + (lane & 15), half = lane >> 4;
#pragma unroll
    for (int r = 0; r < 8; ++r)
      outf[(r + half * 8) * HID + col] = acc[r] + bias[col];
  }
  __syncthreads();
  for (int idx = tid; idx < 16 * HID; idx += 128) {
    int r = idx >> 7, gr = r0 + r;
    if (gr < N) out[(size_t)gr * HID + (idx & 127)] = outf[idx];
  }
}

// ---------------- edge pass 1: kv build, K & V MLPs, logits, seg-max -------
struct EdgeArgs {
  const float* hn;    // node features for kv (N,128)
  const float* x;     // coordinates (N,3)
  const float* ea;    // edge_attr (E,4)
  const int *src, *dst;
  const float* q;     // per-node query (N,128)
  const float *eww, *ewb;
  MLPW wk, wv;
  float* logits;      // (E,16)
  unsigned* mx;       // (N,16) encoded
  void* vout;         // bf16 (E,128) if doutv==128, else f32 (E,16)
  int E, doutv;
};

__global__ __launch_bounds__(128)
void edge_pass1(EdgeArgs ar) {
  __shared__ __bf16 kvt[ETILE * KVPAD];     // 45 KB
  __shared__ float  mid[ETILE * HID];       // 32 KB (layer1 out & final out)
  __shared__ __bf16 midb[ETILE * HID];      // 16 KB
  __shared__ float  df[ETILE * NG];
  __shared__ float  eat[ETILE * EDIM];
  __shared__ float  dist[ETILE], ews[ETILE];
  __shared__ int    ssrc[ETILE], sdst[ETILE];
  const int tid = threadIdx.x, e0 = blockIdx.x * ETILE;

  if (tid < ETILE) {
    int ge = e0 + tid, s = 0, d = 0;
    if (ge < ar.E) { s = ar.src[ge]; d = ar.dst[ge]; }
    ssrc[tid] = s; sdst[tid] = d;
    float dx = ar.x[d * 3 + 0] - ar.x[s * 3 + 0];
    float dy = ar.x[d * 3 + 1] - ar.x[s * 3 + 1];
    float dz = ar.x[d * 3 + 2] - ar.x[s * 3 + 2];
    dist[tid] = sqrtf(dx * dx + dy * dy + dz * dz);
  }
  for (int idx = tid; idx < ETILE * EDIM; idx += 128) {
    int e = idx >> 2, a = idx & 3, ge = e0 + e;
    eat[idx] = (ge < ar.E) ? ar.ea[(size_t)ge * 4 + a] : 0.f;
  }
  __syncthreads();
  // Gaussian smearing
  const float step = 10.f / 19.f;
  const float coeff = -0.5f / (step * step);
  for (int i = tid; i < ETILE * NG; i += 128) {
    int e = i / NG, g = i % NG;
    float dd = dist[e] - (float)g * step;
    df[i] = __expf(coeff * dd * dd);
  }
  __syncthreads();
  // e_w = sigmoid(r_feat . ew_w + ew_b)
  if (tid < ETILE) {
    float acc = ar.ewb[0];
    for (int a = 0; a < 4; ++a)
      for (int g = 0; g < NG; ++g)
        acc += eat[tid * 4 + a] * df[tid * NG + g] * ar.eww[a * NG + g];
    ews[tid] = 1.f / (1.f + __expf(-acc));
  }
  // build kv tile: [ea(4) | r_feat(80) | h[dst](128) | h[src](128) | pad]
  for (int idx = tid; idx < ETILE * KVPAD; idx += 128) {
    int e = idx / KVPAD, c = idx % KVPAD, ge = e0 + e;
    float v = 0.f;
    if (ge < ar.E) {
      if (c < 4) v = eat[e * 4 + c];
      else if (c < 84)  { int k = c - 4; v = eat[e * 4 + k / NG] * df[e * NG + k % NG]; }
      else if (c < 212) v = ar.hn[(size_t)sdst[e] * 128 + (c - 84)];
      else if (c < 340) v = ar.hn[(size_t)ssrc[e] * 128 + (c - 212)];
    }
    kvt[idx] = f2bf(v);
  }
  __syncthreads();
  // K = MLP_k(kv)
  mlp_tile64(kvt, KVPAD, ar.wk, mid, midb, tid);
  // logits + segment max
  for (int p = tid; p < ETILE * 16; p += 128) {
    int e = p >> 4, h = p & 15, ge = e0 + e;
    if (ge < ar.E) {
      int d = sdst[e];
      float s = 0.f;
#pragma unroll
      for (int j = 0; j < HD; ++j)
        s += ar.q[(size_t)d * 128 + h * HD + j] * mid[e * 128 + h * HD + j];
      s *= 0.35355339059327373f;  // 1/sqrt(HD)
      ar.logits[(size_t)ge * 16 + h] = s;
      atomicMax(&ar.mx[(size_t)d * 16 + h], encf(s));
    }
  }
  __syncthreads();
  // V = MLP_v(kv) * e_w
  mlp_tile64(kvt, KVPAD, ar.wv, mid, midb, tid);
  if (ar.doutv == 128) {
    unsigned short* vb = (unsigned short*)ar.vout;
    for (int idx = tid; idx < ETILE * 128; idx += 128) {
      int e = idx >> 7, ge = e0 + e;
      if (ge < ar.E)
        vb[(size_t)ge * 128 + (idx & 127)] = bfbits(mid[idx] * ews[e]);
    }
  } else {
    float* vf = (float*)ar.vout;
    for (int idx = tid; idx < ETILE * 16; idx += 128) {
      int e = idx >> 4, ge = e0 + e;
      if (ge < ar.E)
        vf[(size_t)ge * 16 + (idx & 15)] = mid[e * 16 + (idx & 15)] * ews[e];
    }
  }
}

// ---------------- scatter softmax pass 2: exp + denominator ----------------
__global__ void softmax_pass(float* __restrict__ lg, const unsigned* __restrict__ mx,
                             float* __restrict__ den, const int* __restrict__ dst,
                             int E) {
  long long i = (long long)blockIdx.x * blockDim.x + threadIdx.x;
  if (i < (long long)E * 16) {
    int e = (int)(i >> 4), h = (int)(i & 15);
    int d = dst[e];
    float ex = __expf(lg[i] - decf(mx[(size_t)d * 16 + h]));
    lg[i] = ex;
    atomicAdd(&den[(size_t)d * 16 + h], ex);
  }
}

// ---------------- pass 3: weighted accumulate (x2h) ----------------
__global__ void accum_x2h(const float* __restrict__ lg, const float* __restrict__ den,
                          const unsigned short* __restrict__ vbf,
                          const int* __restrict__ dst, float* __restrict__ outacc,
                          int E) {
  long long i = (long long)blockIdx.x * blockDim.x + threadIdx.x;
  if (i < (long long)E * 128) {
    int e = (int)(i >> 7), c = (int)(i & 127), h = c >> 3;
    int d = dst[e];
    float alpha = lg[(size_t)e * 16 + h] / (den[(size_t)d * 16 + h] + 1e-16f);
    atomicAdd(&outacc[(size_t)d * 128 + c], alpha * bf2f(vbf[i]));
  }
}

// ---------------- pass 3: weighted accumulate (h2x) ----------------
__global__ void accum_h2x(const float* __restrict__ lg, const float* __restrict__ den,
                          const float* __restrict__ v16, const int* __restrict__ src,
                          const int* __restrict__ dst, const float* __restrict__ x,
                          float* __restrict__ xacc, int E) {
  int e = blockIdx.x * blockDim.x + threadIdx.x;
  if (e < E) {
    int d = dst[e], s = src[e];
    float ssum = 0.f;
#pragma unroll
    for (int h = 0; h < 16; ++h)
      ssum += lg[(size_t)e * 16 + h] / (den[(size_t)d * 16 + h] + 1e-16f) *
              v16[(size_t)e * 16 + h];
#pragma unroll
    for (int k = 0; k < 3; ++k)
      atomicAdd(&xacc[(size_t)d * 3 + k], ssum * (x[d * 3 + k] - x[s * 3 + k]));
  }
}

__global__ void scatter_pad(const float* __restrict__ hs, const int* __restrict__ perm,
                            float* __restrict__ pad, int N_sub) {
  long long i = (long long)blockIdx.x * blockDim.x + threadIdx.x;
  if (i < (long long)N_sub * 128) {
    int r = (int)(i >> 7), c = (int)(i & 127);
    pad[(size_t)perm[r] * 128 + c] = hs[i];
  }
}

__global__ void final_x(const float* __restrict__ x, const float* __restrict__ xacc,
                        const int* __restrict__ mask, float* __restrict__ outx, int N) {
  int i = blockIdx.x * blockDim.x + threadIdx.x;
  if (i < N * 3) {
    int r = i / 3;
    outx[i] = x[i] + (xacc[i] * (1.f / 16.f)) * (float)mask[r];
  }
}

// ===========================================================================
extern "C" void kernel_launch(void* const* d_in, const int* in_sizes, int n_in,
                              void* d_out, int out_size, void* d_ws, size_t ws_size,
                              hipStream_t stream) {
  const float* h      = (const float*)d_in[0];
  const float* h_sub  = (const float*)d_in[1];
  const float* x      = (const float*)d_in[2];
  const float* x_sub  = (const float*)d_in[3];
  const float* ea     = (const float*)d_in[4];
  const float* ea_sub = (const float*)d_in[5];
  const float* P      = (const float*)d_in[6];
  const int* ei       = (const int*)d_in[7];
  const int* ei_sub   = (const int*)d_in[8];
  const int* mask     = (const int*)d_in[9];
  const int* perm     = (const int*)d_in[10];

  const int N     = in_sizes[0] / HID;
  const int N_sub = in_sizes[1] / HID;
  const int E     = in_sizes[4] / EDIM;
  const int E_sub = in_sizes[5] / EDIM;
  const int* src   = ei,     *dst   = ei + E;
  const int* src_s = ei_sub, *dst_s = ei_sub + E_sub;
  float* out_h = (float*)d_out;                       // x2h_out (N,128)
  float* out_x = (float*)d_out + (size_t)N * HID;     // x (N,3)

  // ---- workspace bump allocator ----
  char* wsb = (char*)d_ws; size_t wsoff = 0;
  auto alloc = [&](size_t bytes) -> void* {
    void* p = wsb + wsoff; wsoff = (wsoff + bytes + 255) & ~(size_t)255; return p;
  };
  auto packW = [&](size_t woff, int din, int dout) -> unsigned* {
    int chunks = (din + 31) / 32, tiles = dout / 16;
    int total = chunks * tiles * 32 * 8;
    unsigned* dstp = (unsigned*)alloc((size_t)total * 4);
    pack_weight<<<dim3((total + 255) / 256), dim3(256), 0, stream>>>(
        P + woff, din, dout, dstp, chunks);
    return dstp;
  };
  // ---- params flattened in JAX pytree (sorted-key) order ----
  size_t po = 0;
  const size_t off_fus_b = po; po += HID;
  const size_t off_fus_w = po; po += 256 * HID;
  auto consumeMLP = [&](int din, int dout) -> MLPW {
    MLPW m; size_t o = po;
    m.b1 = P + o; o += HID;
    m.b2 = P + o; o += dout;
    m.be = P + o; o += HID;
    m.g  = P + o; o += HID;
    size_t w1 = o; o += (size_t)din * HID;
    size_t w2 = o; o += (size_t)HID * dout;
    m.w1p = packW(w1, din, HID);
    m.w2p = packW(w2, HID, dout);
    m.chunks1 = (din + 31) / 32; m.douttiles = dout / 16;
    po = o; return m;
  };
  // h2x: ew_b, ew_w, xk, xq, xv
  const size_t off_x_ewb = po; po += 1;
  const size_t off_x_eww = po; po += RFEAT;
  MLPW m_xk = consumeMLP(KVDIM, HID);
  MLPW m_xq = consumeMLP(HID, HID);
  MLPW m_xv = consumeMLP(KVDIM, HEADS);
  // x2h: ew_b, ew_w, hk, hq, hv, out
  const size_t off_m_ewb = po; po += 1;
  const size_t off_m_eww = po; po += RFEAT;
  MLPW m_hk  = consumeMLP(KVDIM, HID);
  MLPW m_hq  = consumeMLP(HID, HID);
  MLPW m_hv  = consumeMLP(KVDIM, HID);
  MLPW m_out = consumeMLP(256, HID);
  // x2h_sub
  const size_t off_s_ewb = po; po += 1;
  const size_t off_s_eww = po; po += RFEAT;
  MLPW s_hk  = consumeMLP(KVDIM, HID);
  MLPW s_hq  = consumeMLP(HID, HID);
  MLPW s_hv  = consumeMLP(KVDIM, HID);
  MLPW s_out = consumeMLP(256, HID);
  unsigned* fus_p = packW(off_fus_w, 256, HID);

  // ---- intermediate buffers ----
  float* qm   = (float*)alloc((size_t)N * HID * 4);
  float* qs   = (float*)alloc((size_t)N_sub * HID * 4);
  float* q2   = (float*)alloc((size_t)N * HID * 4);
  unsigned short* v_m = (unsigned short*)alloc((size_t)E * HID * 2);
  unsigned short* v_s = (unsigned short*)alloc((size_t)E_sub * HID * 2);
  float* v_x  = (float*)alloc((size_t)E * HEADS * 4);
  float* lg_m = (float*)alloc((size_t)E * HEADS * 4);
  float* lg_s = (float*)alloc((size_t)E_sub * HEADS * 4);
  float* lg_x = (float*)alloc((size_t)E * HEADS * 4);
  unsigned* mx_m = (unsigned*)alloc((size_t)N * HEADS * 4);
  unsigned* mx_s = (unsigned*)alloc((size_t)N_sub * HEADS * 4);
  unsigned* mx_x = (unsigned*)alloc((size_t)N * HEADS * 4);
  float* den_m = (float*)alloc((size_t)N * HEADS * 4);
  float* den_s = (float*)alloc((size_t)N_sub * HEADS * 4);
  float* den_x = (float*)alloc((size_t)N * HEADS * 4);
  float* acc_m = (float*)alloc((size_t)N * HID * 4);
  float* acc_s = (float*)alloc((size_t)N_sub * HID * 4);
  float* xacc  = (float*)alloc((size_t)N * 3 * 4);
  float* h_in  = (float*)alloc((size_t)N * HID * 4);
  float* h_ins = (float*)alloc((size_t)N_sub * HID * 4);
  float* padb  = (float*)alloc((size_t)N * HID * 4);

  auto fill = [&](void* p, unsigned v, long long n) {
    fill_u32<<<dim3((unsigned)((n + 255) / 256)), dim3(256), 0, stream>>>(
        (unsigned*)p, v, n);
  };
  fill(mx_m, MX_INIT, (long long)N * HEADS);
  fill(mx_s, MX_INIT, (long long)N_sub * HEADS);
  fill(mx_x, MX_INIT, (long long)N * HEADS);
  fill(den_m, 0u, (long long)N * HEADS);
  fill(den_s, 0u, (long long)N_sub * HEADS);
  fill(den_x, 0u, (long long)N * HEADS);
  fill(acc_m, 0u, (long long)N * HID);
  fill(acc_s, 0u, (long long)N_sub * HID);
  fill(xacc, 0u, (long long)N * 3);
  fill(padb, 0u, (long long)N * HID);

  const dim3 blk(128);
  const dim3 gN((N + 15) / 16), gNs((N_sub + 15) / 16);
  const dim3 gE((E + ETILE - 1) / ETILE), gEs((E_sub + ETILE - 1) / ETILE);

  // ================= x2h on main graph =================
  node_mlp<<<gN, blk, 0, stream>>>(h, nullptr, m_hq, nullptr, qm, N);
  {
    EdgeArgs a; a.hn = h; a.x = x; a.ea = ea; a.src = src; a.dst = dst;
    a.q = qm; a.eww = P + off_m_eww; a.ewb = P + off_m_ewb;
    a.wk = m_hk; a.wv = m_hv; a.logits = lg_m; a.mx = mx_m;
    a.vout = (void*)v_m; a.E = E; a.doutv = 128;
    edge_pass1<<<gE, blk, 0, stream>>>(a);
  }
  softmax_pass<<<dim3((unsigned)(((long long)E * 16 + 255) / 256)), dim3(256), 0, stream>>>(
      lg_m, mx_m, den_m, dst, E);
  accum_x2h<<<dim3((unsigned)(((long long)E * 128 + 255) / 256)), dim3(256), 0, stream>>>(
      lg_m, den_m, v_m, dst, acc_m, E);
  node_mlp<<<gN, blk, 0, stream>>>(acc_m, h, m_out, h, h_in, N);

  // ================= x2h on sub graph =================
  node_mlp<<<gNs, blk, 0, stream>>>(h_sub, nullptr, s_hq, nullptr, qs, N_sub);
  {
    EdgeArgs a; a.hn = h_sub; a.x = x_sub; a.ea = ea_sub; a.src = src_s; a.dst = dst_s;
    a.q = qs; a.eww = P + off_s_eww; a.ewb = P + off_s_ewb;
    a.wk = s_hk; a.wv = s_hv; a.logits = lg_s; a.mx = mx_s;
    a.vout = (void*)v_s; a.E = E_sub; a.doutv = 128;
    edge_pass1<<<gEs, blk, 0, stream>>>(a);
  }
  softmax_pass<<<dim3((unsigned)(((long long)E_sub * 16 + 255) / 256)), dim3(256), 0, stream>>>(
      lg_s, mx_s, den_s, dst_s, E_sub);
  accum_x2h<<<dim3((unsigned)(((long long)E_sub * 128 + 255) / 256)), dim3(256), 0, stream>>>(
      lg_s, den_s, v_s, dst_s, acc_s, E_sub);
  node_mlp<<<gNs, blk, 0, stream>>>(acc_s, h_sub, s_out, h_sub, h_ins, N_sub);

  // ================= fusion =================
  scatter_pad<<<dim3((unsigned)(((long long)N_sub * 128 + 255) / 256)), dim3(256), 0, stream>>>(
      h_ins, perm, padb, N_sub);
  fusion_kernel<<<gN, blk, 0, stream>>>(h_in, padb, fus_p, P + off_fus_b, out_h, N);

  // ================= h2x =================
  node_mlp<<<gN, blk, 0, stream>>>(out_h, nullptr, m_xq, nullptr, q2, N);
  {
    EdgeArgs a; a.hn = out_h; a.x = x; a.ea = ea; a.src = src; a.dst = dst;
    a.q = q2; a.eww = P + off_x_eww; a.ewb = P + off_x_ewb;
    a.wk = m_xk; a.wv = m_xv; a.logits = lg_x; a.mx = mx_x;
    a.vout = (void*)v_x; a.E = E; a.doutv = 16;
    edge_pass1<<<gE, blk, 0, stream>>>(a);
  }
  softmax_pass<<<dim3((unsigned)(((long long)E * 16 + 255) / 256)), dim3(256), 0, stream>>>(
      lg_x, mx_x, den_x, dst, E);
  accum_h2x<<<dim3((E + 255) / 256), dim3(256), 0, stream>>>(
      lg_x, den_x, v_x, src, dst, x, xacc, E);
  final_x<<<dim3((N * 3 + 255) / 256), dim3(256), 0, stream>>>(x, xacc, mask, out_x, N);

  (void)n_in; (void)out_size; (void)ws_size;
}
